// HFPSR_43954695307923
// MI455X (gfx1250) — compile-verified, matrix operand order checked by reading, not run on hardware
//
#include <hip/hip_runtime.h>
#include <hip/hip_bf16.h>

// ---------------------------------------------------------------------------
// Swin block pair for MI455X (gfx1250), wave32 + WMMA bf16 (f32 accumulate).
// Memory-bound (~129 GFLOP vs 100MB state @ 23.3 TB/s): fully fused kernels,
// weights staged into LDS per block via CDNA5 async global->LDS copies.
// ---------------------------------------------------------------------------

typedef __attribute__((ext_vector_type(16))) __bf16 v16bf;
typedef __attribute__((ext_vector_type(8)))  float  v8f;

#define NTHREADS 256
#define MLP_NT   512
#define CH    96
#define NHEAD 6
#define HD    16
#define WSZ   8
#define NTOK  64          // tokens per window
#define HDIM  256         // H == W

__device__ __forceinline__ v8f wmma_bf16(v16bf a, v16bf b, v8f c) {
  return __builtin_amdgcn_wmma_f32_16x16x32_bf16(false, a, false, b,
                                                 (short)0, c, false, false);
}

// CDNA5 async global->LDS copy, 16B per lane (ISA 15.18.3 op 98), ASYNCcnt.
__device__ __forceinline__ void async_copy_b128(unsigned lds_byte, const char* g) {
  asm volatile("global_load_async_to_lds_b128 %0, %1, off"
               :: "v"(lds_byte), "v"(g) : "memory");
}
__device__ __forceinline__ void wait_async0() {
  asm volatile("s_wait_asynccnt 0x0" ::: "memory");
}

// K index for element e of a 16-bit A/B fragment (ISA 7.12.2, 16x32 layout):
//   lanes 0-15 : e0..7 -> K=0..7,  e8..15 -> K=16..23
//   lanes16-31 : e0..7 -> K=8..15, e8..15 -> K=24..31
// Row (A) / column (B) index = lane & 15.  Pairs (2e,2e+1) K-consecutive.
__device__ __forceinline__ v16bf load_frag_row(const __bf16* base, int stride,
                                               int row0, int k0, int lane) {
  union { v16bf v; unsigned int u[8]; } f;
  const __bf16* p = base + (size_t)(row0 + (lane & 15)) * stride + k0
                         + ((lane & 16) >> 1);
#pragma unroll
  for (int i = 0; i < 8; ++i) {
    int k = 2 * i + ((i & 4) << 1);   // 0,2,4,6,16,18,20,22
    f.u[i] = *(const unsigned int*)(p + k);
  }
  return f.v;
}

// K limited to 16 (upper half zero) for the 64x64x16 score GEMM; K>=16 lands
// entirely in element pairs i>=4 for every lane.
__device__ __forceinline__ v16bf load_frag_row_k16(const __bf16* base, int stride,
                                                   int row0, int coff, int lane) {
  union { v16bf v; unsigned int u[8]; } f;
  const __bf16* p = base + (size_t)(row0 + (lane & 15)) * stride + coff
                         + ((lane & 16) >> 1);
#pragma unroll
  for (int i = 0; i < 4; ++i) f.u[i] = *(const unsigned int*)(p + 2 * i);
#pragma unroll
  for (int i = 4; i < 8; ++i) f.u[i] = 0u;
  return f.v;
}

// B fragment when memory is [K][N] (V matrix, token-major).
__device__ __forceinline__ v16bf load_frag_col(const __bf16* base, int stride,
                                               int k0, int col0, int lane) {
  v16bf f;
  int c  = col0 + (lane & 15);
  int kl = (lane & 16) >> 1;
#pragma unroll
  for (int e = 0; e < 16; ++e) {
    int k = (e & 7) | ((e & 8) << 1);
    f[e] = base[(size_t)(k0 + kl + k) * stride + c];
  }
  return f;
}

__device__ __forceinline__ int region1d(int g, int shift) {
  if (shift == 0) return 0;
  if (g < HDIM - WSZ)   return 0;
  if (g < HDIM - shift) return 1;
  return 2;
}

// ---------------------------------------------------------------------------
// Prep kernels: weight f32->bf16, relative-position-bias gather.
// ---------------------------------------------------------------------------
__global__ void cvt_bf16_kernel(const float* __restrict__ src,
                                __bf16* __restrict__ dst, int n) {
  int i = blockIdx.x * NTHREADS + threadIdx.x;
  if (i < n) dst[i] = (__bf16)src[i];
}

__global__ void gather_bias_kernel(const float* __restrict__ rpb,
                                   float* __restrict__ out) {
  // out[layer][head][n][m], rpb[layer][225][6]
  int i = blockIdx.x * NTHREADS + threadIdx.x;
  if (i >= 2 * NHEAD * NTOK * NTOK) return;
  int m = i & 63, n = (i >> 6) & 63;
  int h = (i >> 12) % NHEAD, l = i / (NHEAD * 4096);
  int rn = n >> 3, cn = n & 7, rm = m >> 3, cm = m & 7;
  int rel = (rn - rm + WSZ - 1) * (2 * WSZ - 1) + (cn - cm + WSZ - 1);
  out[i] = rpb[(l * 225 + rel) * NHEAD + h];
}

// ---------------------------------------------------------------------------
// Fused window attention. One block (8 waves) per 8x8 window.
// Dynamic LDS layout (147456 B, 2 blocks/WGP at 320KB):
//   [0      , 36864) qkv   bf16 [64][288]
//   [36864  , 53248) score f32  [64][64]   (reused: xn bf16 [64][96] early)
//   [53248  , 61440) P     bf16 [64][64]
//   [61440  , 73728) O     bf16 [64][96]
//   [73728  ,129024) wqkv  bf16 [288][96]  (async-staged)
//   [129024 ,147456) wproj bf16 [96][96]   (async-staged)
// ---------------------------------------------------------------------------
__global__ __launch_bounds__(NTHREADS) void attn_kernel(
    const float* xin, float* xout,
    const __bf16* __restrict__ wqkv, const float* __restrict__ bqkv,
    const __bf16* __restrict__ wproj, const float* __restrict__ bproj,
    const float* __restrict__ biasTab,
    const float* __restrict__ n1w, const float* __restrict__ n1b,
    int shift) {
  extern __shared__ char smem[];
  __bf16* s_qkv  = (__bf16*)smem;                    // 64 x 288
  float*  s_sc   = (float*)(smem + 36864);           // 64 x 64  f32
  __bf16* s_xn   = (__bf16*)(smem + 36864);          // 64 x 96  (overlaps s_sc)
  __bf16* s_p    = (__bf16*)(smem + 53248);          // 64 x 64
  __bf16* s_o    = (__bf16*)(smem + 61440);          // 64 x 96
  __bf16* s_wqkv = (__bf16*)(smem + 73728);          // 288 x 96
  __bf16* s_wprj = (__bf16*)(smem + 129024);         // 96 x 96

  const int tid  = threadIdx.x;
  const int lane = tid & 31;
  const int wave = tid >> 5;
  const int w    = blockIdx.x;
  const int bimg = w >> 10;
  const int wh   = (w >> 5) & 31;
  const int ww   = w & 31;

  // ---- stage this layer's weights into LDS (async, overlapped with LN) ----
  {
    unsigned sbase = (unsigned)(uintptr_t)smem + 73728u;
    for (int c = tid; c < 4608; c += NTHREADS) {   // 4608 x 16B = 73728 B
      const char* src = (c < 3456) ? (const char*)wqkv + c * 16
                                   : (const char*)wproj + (c - 3456) * 16;
      async_copy_b128(sbase + (unsigned)c * 16u, src);
    }
  }

  // ---- LayerNorm1 on this window's 64 tokens (gather with cyclic shift) ----
  if (tid < NTOK) {
    int rt = tid >> 3, ct = tid & 7;
    int orr = (wh * WSZ + rt + shift) & (HDIM - 1);
    int occ = (ww * WSZ + ct + shift) & (HDIM - 1);
    const float* row = xin + ((size_t)bimg * 65536 + orr * HDIM + occ) * CH;
    float s = 0.f, ss = 0.f;
    for (int c = 0; c < CH; ++c) { float v = row[c]; s += v; ss += v * v; }
    float mean = s * (1.f / CH);
    float var  = ss * (1.f / CH) - mean * mean;
    float rstd = rsqrtf(var + 1e-5f);
    for (int c = 0; c < CH; ++c)
      s_xn[tid * CH + c] = (__bf16)((row[c] - mean) * rstd * n1w[c] + n1b[c]);
  }
  wait_async0();
  __syncthreads();

  // ---- QKV GEMM: (64 x 96) @ (96 x 288) -> LDS, q pre-scaled by HD^-0.5 ----
  for (int tile = wave; tile < 4 * 18; tile += 8) {
    int mt = tile / 18, nt = tile % 18;
    v8f acc = {};
#pragma unroll
    for (int k0 = 0; k0 < CH; k0 += 32) {
      v16bf a = load_frag_row(s_xn, CH, mt * 16, k0, lane);
      v16bf b = load_frag_row(s_wqkv, CH, nt * 16, k0, lane);
      acc = wmma_bf16(a, b, acc);
    }
    int col = nt * 16 + (lane & 15);
    float bb = bqkv[col];
    float sc = (col < CH) ? 0.25f : 1.f;   // SCALE = 16^-0.5
#pragma unroll
    for (int r = 0; r < 8; ++r) {
      int m = mt * 16 + r + ((lane & 16) >> 1);
      s_qkv[m * 288 + col] = (__bf16)((acc[r] + bb) * sc);
    }
  }
  __syncthreads();

  // ---- per-head attention ----
  for (int h = 0; h < NHEAD; ++h) {
    // scores = q @ k^T (K=16 via zero-padded K=32 frags) + bias + mask
    for (int tile = wave; tile < 16; tile += 8) {
      int mt = tile >> 2, nt = tile & 3;
      v16bf a = load_frag_row_k16(s_qkv, 288, mt * 16, h * HD, lane);
      v16bf b = load_frag_row_k16(s_qkv, 288, nt * 16, CH + h * HD, lane);
      v8f acc = {};
      acc = wmma_bf16(a, b, acc);
      int colT = nt * 16 + (lane & 15);
      int ridc = 3 * region1d(wh * WSZ + (colT >> 3), shift)
                   + region1d(ww * WSZ + (colT & 7), shift);
#pragma unroll
      for (int r = 0; r < 8; ++r) {
        int rowT = mt * 16 + r + ((lane & 16) >> 1);
        int ridr = 3 * region1d(wh * WSZ + (rowT >> 3), shift)
                     + region1d(ww * WSZ + (rowT & 7), shift);
        float v = acc[r] + biasTab[(h * NTOK + rowT) * NTOK + colT];
        if (ridr != ridc) v -= 100.f;
        s_sc[rowT * NTOK + colT] = v;
      }
    }
    __syncthreads();

    // softmax over each row (one thread per row)
    if (tid < NTOK) {
      float mx = -1e30f;
      for (int m = 0; m < NTOK; ++m) mx = fmaxf(mx, s_sc[tid * NTOK + m]);
      float sum = 0.f;
      for (int m = 0; m < NTOK; ++m) {
        float e = __expf(s_sc[tid * NTOK + m] - mx);
        s_sc[tid * NTOK + m] = e;
        sum += e;
      }
      float inv = 1.f / sum;
      for (int m = 0; m < NTOK; ++m)
        s_p[tid * NTOK + m] = (__bf16)(s_sc[tid * NTOK + m] * inv);
    }
    __syncthreads();

    // O_h = P @ V_h : (64 x 64) @ (64 x 16)
    if (wave < 4) {
      int mt = wave;
      v8f acc = {};
#pragma unroll
      for (int k0 = 0; k0 < NTOK; k0 += 32) {
        v16bf a = load_frag_row(s_p, NTOK, mt * 16, k0, lane);
        v16bf b = load_frag_col(s_qkv, 288, k0, 2 * CH + h * HD, lane);
        acc = wmma_bf16(a, b, acc);
      }
      int col = h * HD + (lane & 15);
#pragma unroll
      for (int r = 0; r < 8; ++r) {
        int m = mt * 16 + r + ((lane & 16) >> 1);
        s_o[m * CH + col] = (__bf16)acc[r];
      }
    }
    __syncthreads();
  }

  // ---- output projection + residual, scatter back with reverse roll ----
  for (int tile = wave; tile < 4 * 6; tile += 8) {
    int mt = tile / 6, nt = tile % 6;
    v8f acc = {};
#pragma unroll
    for (int k0 = 0; k0 < CH; k0 += 32) {
      v16bf a = load_frag_row(s_o, CH, mt * 16, k0, lane);
      v16bf b = load_frag_row(s_wprj, CH, nt * 16, k0, lane);
      acc = wmma_bf16(a, b, acc);
    }
    int col = nt * 16 + (lane & 15);
    float bb = bproj[col];
#pragma unroll
    for (int r = 0; r < 8; ++r) {
      int t   = mt * 16 + r + ((lane & 16) >> 1);
      int orr = (wh * WSZ + (t >> 3) + shift) & (HDIM - 1);
      int occ = (ww * WSZ + (t & 7) + shift) & (HDIM - 1);
      size_t idx = ((size_t)bimg * 65536 + orr * HDIM + occ) * CH + col;
      xout[idx] = xin[idx] + acc[r] + bb;
    }
  }
}

// ---------------------------------------------------------------------------
// Fused MLP: LN2 -> fc1 -> GELU -> fc2 -> residual. 128 tokens per block,
// 512 threads (16 waves).  Dynamic LDS (270336 B, 1 block/WGP):
//   [0      , 24576) xn bf16 [128][96]
//   [24576  ,122880) y  bf16 [128][384]
//   [122880 ,196608) w1 bf16 [384][96]  (async-staged)
//   [196608 ,270336) w2 bf16 [96][384]  (async-staged)
// ---------------------------------------------------------------------------
__global__ __launch_bounds__(MLP_NT) void mlp_kernel(
    float* x,
    const __bf16* __restrict__ w1, const float* __restrict__ b1,
    const __bf16* __restrict__ w2, const float* __restrict__ b2,
    const float* __restrict__ n2w, const float* __restrict__ n2b) {
  extern __shared__ char smem[];
  __bf16* s_xn = (__bf16*)smem;              // 128 x 96
  __bf16* s_y  = (__bf16*)(smem + 24576);    // 128 x 384
  __bf16* s_w1 = (__bf16*)(smem + 122880);   // 384 x 96
  __bf16* s_w2 = (__bf16*)(smem + 196608);   // 96 x 384

  const int tid = threadIdx.x, lane = tid & 31, wave = tid >> 5;
  const size_t base = (size_t)blockIdx.x * 128;

  // stage w1+w2 (9216 x 16B = 147456 B), overlapped with LN
  {
    unsigned sbase = (unsigned)(uintptr_t)smem + 122880u;
    for (int c = tid; c < 9216; c += MLP_NT) {
      const char* src = (c < 4608) ? (const char*)w1 + c * 16
                                   : (const char*)w2 + (c - 4608) * 16;
      async_copy_b128(sbase + (unsigned)c * 16u, src);
    }
  }

  if (tid < 128) {
    const float* row = x + (base + tid) * CH;
    float s = 0.f, ss = 0.f;
    for (int c = 0; c < CH; ++c) { float v = row[c]; s += v; ss += v * v; }
    float mean = s * (1.f / CH);
    float var  = ss * (1.f / CH) - mean * mean;
    float rstd = rsqrtf(var + 1e-5f);
    for (int c = 0; c < CH; ++c)
      s_xn[tid * CH + c] = (__bf16)((row[c] - mean) * rstd * n2w[c] + n2b[c]);
  }
  wait_async0();
  __syncthreads();

  // fc1 + exact GELU: (128 x 96) @ (96 x 384); 8x24 tiles over 16 waves
  for (int tile = wave; tile < 8 * 24; tile += 16) {
    int mt = tile / 24, nt = tile % 24;
    v8f acc = {};
#pragma unroll
    for (int k0 = 0; k0 < CH; k0 += 32) {
      v16bf a = load_frag_row(s_xn, CH, mt * 16, k0, lane);
      v16bf b = load_frag_row(s_w1, CH, nt * 16, k0, lane);
      acc = wmma_bf16(a, b, acc);
    }
    int col = nt * 16 + (lane & 15);
    float bb = b1[col];
#pragma unroll
    for (int r = 0; r < 8; ++r) {
      float v = acc[r] + bb;
      v = 0.5f * v * (1.f + erff(v * 0.70710678118654752f));
      int m = mt * 16 + r + ((lane & 16) >> 1);
      s_y[m * 384 + col] = (__bf16)v;
    }
  }
  __syncthreads();

  // fc2 + residual: (128 x 384) @ (384 x 96); 8x6 tiles over 16 waves
  for (int tile = wave; tile < 8 * 6; tile += 16) {
    int mt = tile / 6, nt = tile % 6;
    v8f acc = {};
#pragma unroll
    for (int k0 = 0; k0 < 384; k0 += 32) {
      v16bf a = load_frag_row(s_y, 384, mt * 16, k0, lane);
      v16bf b = load_frag_row(s_w2, 384, nt * 16, k0, lane);
      acc = wmma_bf16(a, b, acc);
    }
    int col = nt * 16 + (lane & 15);
    float bb = b2[col];
#pragma unroll
    for (int r = 0; r < 8; ++r) {
      size_t idx = (base + mt * 16 + r + ((lane & 16) >> 1)) * CH + col;
      x[idx] += acc[r] + bb;
    }
  }
}

// ---------------------------------------------------------------------------
extern "C" void kernel_launch(void* const* d_in, const int* in_sizes, int n_in,
                              void* d_out, int out_size, void* d_ws, size_t ws_size,
                              hipStream_t stream) {
  const float* x     = (const float*)d_in[0];
  const float* qkvw  = (const float*)d_in[1];
  const float* qkvb  = (const float*)d_in[2];
  const float* projw = (const float*)d_in[3];
  const float* projb = (const float*)d_in[4];
  const float* rpb   = (const float*)d_in[5];
  const float* n1w   = (const float*)d_in[6];
  const float* n1b   = (const float*)d_in[7];
  const float* n2w   = (const float*)d_in[8];
  const float* n2b   = (const float*)d_in[9];
  const float* f1w   = (const float*)d_in[10];
  const float* f1b   = (const float*)d_in[11];
  const float* f2w   = (const float*)d_in[12];
  const float* f2b   = (const float*)d_in[13];

  // workspace: bf16 weights + gathered bias (all L2-resident)
  char* ws = (char*)d_ws;
  __bf16* wqkv  = (__bf16*)(ws);                        // 2*288*96
  __bf16* wproj = (__bf16*)(ws + 110592);               // 2*96*96
  __bf16* wfc1  = (__bf16*)(ws + 110592 + 36864);       // 2*384*96
  __bf16* wfc2  = (__bf16*)(ws + 110592 + 36864 + 147456);
  float*  bias  = (float*)(ws + 110592 + 36864 + 147456 + 147456);

  cvt_bf16_kernel<<<(55296 + 255) / 256, 256, 0, stream>>>(qkvw, wqkv, 55296);
  cvt_bf16_kernel<<<(18432 + 255) / 256, 256, 0, stream>>>(projw, wproj, 18432);
  cvt_bf16_kernel<<<(73728 + 255) / 256, 256, 0, stream>>>(f1w, wfc1, 73728);
  cvt_bf16_kernel<<<(73728 + 255) / 256, 256, 0, stream>>>(f2w, wfc2, 73728);
  gather_bias_kernel<<<(49152 + 255) / 256, 256, 0, stream>>>(rpb, bias);

  float* X = (float*)d_out;   // layer state lives in d_out (fully overwritten)
  for (int l = 0; l < 2; ++l) {
    int shift = l ? (WSZ / 2) : 0;
    const float* xi = (l == 0) ? x : X;
    attn_kernel<<<4096, NTHREADS, 147456, stream>>>(
        xi, X,
        wqkv + l * 27648, qkvb + l * 288,
        wproj + l * 9216, projb + l * 96,
        bias + l * 24576,
        n1w + l * 96, n1b + l * 96, shift);
    mlp_kernel<<<2048, MLP_NT, 270336, stream>>>(
        X,
        wfc1 + l * 36864, f1b + l * 384,
        wfc2 + l * 36864, f2b + l * 96,
        n2w + l * 96, n2b + l * 96);
  }
}
